// LinearAttention_1039382086055
// MI455X (gfx1250) — compile-verified, matrix operand order checked by reading, not run on hardware
//
#include <hip/hip_runtime.h>
#include <hip/hip_bf16.h>

#define DIM  256
#define NVOX 8192
#define SCALE 0.125f
#define AW 8   // waves per block in attention kernel

typedef __bf16 bf16;
typedef bf16 v16bf __attribute__((ext_vector_type(16)));
typedef bf16 v8bf  __attribute__((ext_vector_type(8)));
typedef float v8f  __attribute__((ext_vector_type(8)));

// ---------------------------------------------------------------------------
// WMMA helper (CDNA5 wave32, V_WMMA_F32_16X16X32_BF16)
// ---------------------------------------------------------------------------
__device__ __forceinline__ v8f wmma_bf16(v16bf a, v16bf b, v8f c) {
  return __builtin_amdgcn_wmma_f32_16x16x32_bf16(false, a, false, b, (short)0, c,
                                                 false, false);
}

// ---------------------------------------------------------------------------
// CDNA5 async copy: global -> LDS, 16B per lane, tracked by ASYNCcnt.
// VDST operand = 32-bit LDS byte offset (low 32 bits of a generic LDS pointer,
// per ISA 10.2 aperture mapping), VADDR = 64-bit global address.
// ---------------------------------------------------------------------------
__device__ __forceinline__ void async_copy_b128(const bf16* gsrc, bf16* lds_dst) {
  unsigned int l = (unsigned int)(unsigned long long)lds_dst;
  asm volatile("global_load_async_to_lds_b128 %0, %1, off"
               :: "v"(l), "v"(gsrc) : "memory");
}
__device__ __forceinline__ void wait_async_le4() {
  asm volatile("s_wait_asynccnt 0x4" ::: "memory");
}
__device__ __forceinline__ void wait_async_0() {
  asm volatile("s_wait_asynccnt 0x0" ::: "memory");
}

// A-operand loader per ISA 7.12.2 "16-bit A-Matrix 16x32":
//   lane l: row m = l&15, half h = l>>4
//   elements j=0..7  -> K = k0 + 8h + j ; j=8..15 -> K = k0 + 16 + 8h + (j-8)
// B-operand assumed to mirror A with M<->N swapped, so the same loader reads B
// from a row-major [16 x K] (B^T) view: exactly right for Q@K^T and the
// pre-transposed Wt / Vt.
__device__ __forceinline__ v16bf load_tile16(const bf16* __restrict__ base,
                                             int row0, int ld, int k0, int lane) {
  const int m = lane & 15, h = lane >> 4;
  const bf16* p = base + (size_t)(row0 + m) * ld + k0 + 8 * h;
  v8bf lo = *(const v8bf*)(p);
  v8bf hi = *(const v8bf*)(p + 16);
  v16bf r;
#pragma unroll
  for (int j = 0; j < 8; ++j) { r[j] = lo[j]; r[j + 8] = hi[j]; }
  return r;
}

// Reductions across the 16 lanes of a half-wave (xor masks < 16 stay in-group)
__device__ __forceinline__ float half_max(float v) {
  v = fmaxf(v, __shfl_xor(v, 1, 32));
  v = fmaxf(v, __shfl_xor(v, 2, 32));
  v = fmaxf(v, __shfl_xor(v, 4, 32));
  v = fmaxf(v, __shfl_xor(v, 8, 32));
  return v;
}
__device__ __forceinline__ float half_sum(float v) {
  v += __shfl_xor(v, 1, 32);
  v += __shfl_xor(v, 2, 32);
  v += __shfl_xor(v, 4, 32);
  v += __shfl_xor(v, 8, 32);
  return v;
}

// ---------------------------------------------------------------------------
// Prep: f32 -> bf16 conversions
// ---------------------------------------------------------------------------
__global__ void cvt_x_kernel(const float* __restrict__ x, bf16* __restrict__ xb,
                             int n) {
  int i = blockIdx.x * blockDim.x + threadIdx.x;
  if (i < n) xb[i] = (bf16)x[i];
}

__global__ void cvt_w_kernel(const float* __restrict__ Wq,
                             const float* __restrict__ Wk,
                             const float* __restrict__ Wv,
                             const float* __restrict__ Wo,
                             bf16* __restrict__ Wt) {
  int i = blockIdx.x * blockDim.x + threadIdx.x;
  if (i >= 4 * DIM * DIM) return;
  int w = i >> 16;
  int n = (i >> 8) & 255;  // out channel
  int k = i & 255;         // in channel
  const float* W = (w == 0) ? Wq : (w == 1) ? Wk : (w == 2) ? Wv : Wo;
  Wt[(size_t)i] = (bf16)W[(size_t)k * DIM + n];
}

// ---------------------------------------------------------------------------
// QKV projection: Q,K row-major bf16 [N,256]; V transposed bf16 [256,N]
// ---------------------------------------------------------------------------
__global__ void qkv_kernel(const bf16* __restrict__ xb, const bf16* __restrict__ Wt,
                           const float* __restrict__ bq, const float* __restrict__ bk,
                           const float* __restrict__ bv,
                           bf16* __restrict__ Qb, bf16* __restrict__ Kb,
                           bf16* __restrict__ Vt) {
  const int lane = threadIdx.x & 31;
  const int wave = threadIdx.x >> 5;
  const int t = blockIdx.x * (blockDim.x >> 5) + wave;  // 0 .. 3*8192-1
  const int mat = t / 8192;
  const int tid = t - mat * 8192;
  const int rb = (tid >> 4) * 16;
  const int ct = tid & 15;
  const bf16* W = Wt + (size_t)mat * DIM * DIM;

  v8f acc = {};
#pragma unroll
  for (int kt = 0; kt < 8; ++kt) {
    v16bf a = load_tile16(xb, rb, DIM, kt * 32, lane);
    v16bf b = load_tile16(W, ct * 16, DIM, kt * 32, lane);
    acc = wmma_bf16(a, b, acc);
  }
  const float* bias = (mat == 0) ? bq : (mat == 1) ? bk : bv;
  const int n = lane & 15, h = lane >> 4;
  const float bb = bias[ct * 16 + n];
  if (mat == 2) {
#pragma unroll
    for (int r = 0; r < 8; ++r) {
      int row = rb + 8 * h + r;
      Vt[(size_t)(ct * 16 + n) * NVOX + row] = (bf16)(acc[r] + bb);
    }
  } else {
    bf16* O = (mat == 0) ? Qb : Kb;
#pragma unroll
    for (int r = 0; r < 8; ++r) {
      int row = rb + 8 * h + r;
      O[(size_t)row * DIM + ct * 16 + n] = (bf16)(acc[r] + bb);
    }
  }
}

// ---------------------------------------------------------------------------
// Fused attention. Block = 8 waves, each owning one 16-row query block; all
// waves share a double-buffered 32-row K slab staged into LDS with
// global_load_async_to_lds_b128 (overlapped with WMMA on the current slab).
//   Pass 1: streaming softmax stats (row max / sum).
//   Pass 2: recompute S, write normalized attn once, stage P through LDS
//           (C-layout -> A-layout) and accumulate P @ V.
// ---------------------------------------------------------------------------
__global__ void attn_kernel(const bf16* __restrict__ Qb, const bf16* __restrict__ Kb,
                            const bf16* __restrict__ Vt,
                            float* __restrict__ attn, float* __restrict__ of) {
  __shared__ __align__(16) bf16 kbuf[2][32 * DIM];   // 2 x 16 KB K slabs
  __shared__ __align__(16) bf16 pstage[AW][16 * 32]; // per-wave P staging

  const int tid = threadIdx.x;
  const int lane = tid & 31;
  const int wave = tid >> 5;
  const int row0 = (blockIdx.x * AW + wave) * 16;
  const int n = lane & 15, h = lane >> 4;
  const int NJP = NVOX / 32;

  // Stage helper: copy K rows [j0, j0+32) (16 KB contiguous) into buf.
  // 256 threads x 4 x 16B async copies.
  auto stage_k = [&](bf16* buf, int j0) {
    const bf16* src = Kb + (size_t)j0 * DIM;
#pragma unroll
    for (int i = 0; i < 4; ++i) {
      int off = (tid + i * 256) * 8;  // bf16 elements, 8 = 16B
      async_copy_b128(src + off, buf + off);
    }
  };

  // Q tiles for this wave (8 x v16bf = 8 KB / wave), resident for pass 1.
  v16bf qreg[8];
#pragma unroll
  for (int kt = 0; kt < 8; ++kt) qreg[kt] = load_tile16(Qb, row0, DIM, kt * 32, lane);

  float mrun[8], lrun[8];
#pragma unroll
  for (int r = 0; r < 8; ++r) { mrun[r] = -3.0e38f; lrun[r] = 0.0f; }

  // ---- Pass 1: row max / sum ----
  stage_k(&kbuf[0][0], 0);
  for (int jp = 0; jp < NJP; ++jp) {
    const int b = jp & 1;
    if (jp + 1 < NJP) { stage_k(&kbuf[1 - b][0], (jp + 1) * 32); wait_async_le4(); }
    else              { wait_async_0(); }
    __syncthreads();
    const bf16* kb = &kbuf[b][0];
#pragma unroll
    for (int sub = 0; sub < 2; ++sub) {
      v8f acc = {};
#pragma unroll
      for (int kt = 0; kt < 8; ++kt)
        acc = wmma_bf16(qreg[kt], load_tile16(kb, sub * 16, DIM, kt * 32, lane), acc);
#pragma unroll
      for (int r = 0; r < 8; ++r) {
        float s = acc[r] * SCALE;
        float mnew = fmaxf(mrun[r], half_max(s));
        float p = __expf(s - mnew);
        float ts = half_sum(p);
        lrun[r] = lrun[r] * __expf(mrun[r] - mnew) + ts;
        mrun[r] = mnew;
      }
    }
    __syncthreads();
  }
  float rinv[8];
#pragma unroll
  for (int r = 0; r < 8; ++r) rinv[r] = 1.0f / lrun[r];

  // ---- Pass 2: write attn + accumulate P @ V ----
  v8f zero = {};
  v8f oacc[16];
#pragma unroll
  for (int c = 0; c < 16; ++c) oacc[c] = zero;

  bf16* st = &pstage[wave][0];
  stage_k(&kbuf[0][0], 0);
  for (int jp = 0; jp < NJP; ++jp) {
    const int b = jp & 1;
    if (jp + 1 < NJP) { stage_k(&kbuf[1 - b][0], (jp + 1) * 32); wait_async_le4(); }
    else              { wait_async_0(); }
    __syncthreads();
    const bf16* kb = &kbuf[b][0];
#pragma unroll
    for (int sub = 0; sub < 2; ++sub) {
      const int jt = jp * 2 + sub;
      v8f acc = {};
#pragma unroll
      for (int kt = 0; kt < 8; ++kt) {
        v16bf a = load_tile16(Qb, row0, DIM, kt * 32, lane);  // L0-resident
        acc = wmma_bf16(a, load_tile16(kb, sub * 16, DIM, kt * 32, lane), acc);
      }
#pragma unroll
      for (int r = 0; r < 8; ++r) {
        float p = __expf(acc[r] * SCALE - mrun[r]) * rinv[r];
        attn[(size_t)(row0 + 8 * h + r) * NVOX + jt * 16 + n] = p;
        st[(8 * h + r) * 32 + sub * 16 + n] = (bf16)p;  // row-major 16x32 P
      }
    }
    // P (16x32, A from LDS) @ V (32 x 256, B from transposed Vt)
    v16bf ap = load_tile16(st, 0, 32, 0, lane);
#pragma unroll
    for (int c = 0; c < 16; ++c) {
      v16bf bv_ = load_tile16(Vt, c * 16, NVOX, jp * 32, lane);
      oacc[c] = wmma_bf16(ap, bv_, oacc[c]);
    }
    __syncthreads();
  }
#pragma unroll
  for (int c = 0; c < 16; ++c)
#pragma unroll
    for (int r = 0; r < 8; ++r)
      of[(size_t)(row0 + 8 * h + r) * DIM + c * 16 + n] = oacc[c][r];
}

// ---------------------------------------------------------------------------
// Output projection: out = x + ((x - out_feats) @ Wo + bo)
// ---------------------------------------------------------------------------
__global__ void out_kernel(const float* __restrict__ x, const float* __restrict__ of,
                           const bf16* __restrict__ Wot, const float* __restrict__ bo,
                           float* __restrict__ out) {
  const int lane = threadIdx.x & 31;
  const int wave = threadIdx.x >> 5;
  const int t = blockIdx.x * (blockDim.x >> 5) + wave;  // 0..8191
  const int rb = (t >> 4) * 16;
  const int ct = t & 15;
  const int n = lane & 15, h = lane >> 4, m = lane & 15;

  v8f acc = {};
#pragma unroll
  for (int kt = 0; kt < 8; ++kt) {
    const float* px = x + (size_t)(rb + m) * DIM + kt * 32 + 8 * h;
    const float* pf = of + (size_t)(rb + m) * DIM + kt * 32 + 8 * h;
    v16bf a;
#pragma unroll
    for (int j = 0; j < 8; ++j) {
      a[j]     = (bf16)(px[j]      - pf[j]);
      a[j + 8] = (bf16)(px[j + 16] - pf[j + 16]);
    }
    v16bf b = load_tile16(Wot, ct * 16, DIM, kt * 32, lane);
    acc = wmma_bf16(a, b, acc);
  }
  const float bb = bo[ct * 16 + n];
#pragma unroll
  for (int r = 0; r < 8; ++r) {
    int row = rb + 8 * h + r;
    int col = ct * 16 + n;
    out[(size_t)row * DIM + col] = x[(size_t)row * DIM + col] + acc[r] + bb;
  }
}

// ---------------------------------------------------------------------------
extern "C" void kernel_launch(void* const* d_in, const int* in_sizes, int n_in,
                              void* d_out, int out_size, void* d_ws, size_t ws_size,
                              hipStream_t stream) {
  const float* x  = (const float*)d_in[0];
  const float* Wq = (const float*)d_in[1];
  const float* bq = (const float*)d_in[2];
  const float* Wk = (const float*)d_in[3];
  const float* bk = (const float*)d_in[4];
  const float* Wv = (const float*)d_in[5];
  const float* bv = (const float*)d_in[6];
  const float* Wo = (const float*)d_in[7];
  const float* bo = (const float*)d_in[8];

  float* out_final = (float*)d_out;                       // [N, DIM]
  float* attn      = (float*)d_out + (size_t)NVOX * DIM;  // [N, N]

  char* p = (char*)d_ws;
  bf16* xb = (bf16*)p;  p += (size_t)NVOX * DIM * sizeof(bf16);
  bf16* Qb = (bf16*)p;  p += (size_t)NVOX * DIM * sizeof(bf16);
  bf16* Kb = (bf16*)p;  p += (size_t)NVOX * DIM * sizeof(bf16);
  bf16* Vt = (bf16*)p;  p += (size_t)DIM * NVOX * sizeof(bf16);
  bf16* Wt = (bf16*)p;  p += (size_t)4 * DIM * DIM * sizeof(bf16);
  float* of = (float*)p;  // [N, DIM] f32

  cvt_x_kernel<<<(NVOX * DIM) / 256, 256, 0, stream>>>(x, xb, NVOX * DIM);
  cvt_w_kernel<<<(4 * DIM * DIM) / 256, 256, 0, stream>>>(Wq, Wk, Wv, Wo, Wt);
  qkv_kernel<<<(3 * 8192) / 8, 256, 0, stream>>>(xb, Wt, bq, bk, bv, Qb, Kb, Vt);
  attn_kernel<<<(NVOX / 16) / AW, 256, 0, stream>>>(Qb, Kb, Vt, attn, of);
  out_kernel<<<8192 / 8, 256, 0, stream>>>(x, of, Wt + (size_t)3 * DIM * DIM, bo,
                                           out_final);
}
